// NeighborhoodAttentionAggregator_6923487281771
// MI455X (gfx1250) — matrix operand order repaired and checked by reading, not compiled, not run on hardware
//
#include <hip/hip_runtime.h>

typedef __attribute__((ext_vector_type(4)))  float  v4f;
typedef __attribute__((ext_vector_type(8)))  float  v8f;
typedef __attribute__((ext_vector_type(4)))  __bf16 v4bf;
typedef __attribute__((ext_vector_type(8)))  __bf16 v8bf;
typedef __attribute__((ext_vector_type(16))) __bf16 v16bf;

#define SN 264   // bf16 row stride (256 data + 8 pad)
#define SC 260   // f32 row stride (256 data + 4 pad)
#define SKsz 66  // f32 row stride for k/q results (64 data + 2 pad)
#define SL 17

// byte offsets in dynamic LDS (all 16B aligned)
#define OFF_SN    0        // neigh bf16   [256][SN]   135168 B
#define OFF_SCB   135168   // center bf16  [16][SN]      8448 B
#define OFF_SCXB  143616   // context bf16 [16][SN]      8448 B
#define OFF_SC    152064   // center f32   [16][SC]     16640 B
#define OFF_SCX   168704   // context f32  [16][SC]     16640 B
#define OFF_SK    185344   // k result f32 [256][SKsz]  67584 B
#define OFF_SQ    252928   // q result f32 [16][SKsz]    4224 B
#define OFF_SL    257152   // logits/attn  [16][SL]      1088 B
#define SMEM_BYTES 258240

static __device__ __forceinline__ v16bf jn(v8bf lo, v8bf hi) {
  return __builtin_shufflevector(lo, hi, 0,1,2,3,4,5,6,7,8,9,10,11,12,13,14,15);
}
static __device__ __forceinline__ v8bf ldb8(const __bf16* p) { return *(const v8bf*)p; }

#define BF16WMMA(af, bf, c) \
  __builtin_amdgcn_wmma_f32_16x16x32_bf16(false, (af), false, (bf), (short)0, (c), false, false)

// ---- pre-pass: fp32 weights -> bf16 workspace ----
// ws layout (elements): Wq[16384] | Wk[16384] | Wg[131072]
__global__ void na_weights_to_bf16(const float* __restrict__ wq,
                                   const float* __restrict__ wk,
                                   const float* __restrict__ wg,
                                   __bf16* __restrict__ o) {
  int i = blockIdx.x * 256 + threadIdx.x;
  if (i < 16384)       o[i] = (__bf16)wq[i];
  else if (i < 32768)  o[i] = (__bf16)wk[i - 16384];
  else if (i < 163840) o[i] = (__bf16)wg[i - 32768];
}

__global__ __launch_bounds__(256) void na_fused(
    const float* __restrict__ center,  // [B,256]
    const float* __restrict__ neigh,   // [B,16,256]
    const float* __restrict__ nwt,     // [B,16]
    const int*   __restrict__ vmask,   // [B]
    const float* __restrict__ bq,      // [64]
    const float* __restrict__ bk,      // [64]
    const float* __restrict__ bg,      // [256]
    const __bf16* __restrict__ wsb,    // bf16 weights in workspace
    float* __restrict__ out)           // [B,256]
{
  extern __shared__ char sm[];
  __bf16* sN   = (__bf16*)(sm + OFF_SN);
  __bf16* sCb  = (__bf16*)(sm + OFF_SCB);
  __bf16* sCxb = (__bf16*)(sm + OFF_SCXB);
  float*  sC   = (float*)(sm + OFF_SC);
  float*  sCx  = (float*)(sm + OFF_SCX);
  float*  sK   = (float*)(sm + OFF_SK);
  float*  sQ   = (float*)(sm + OFF_SQ);
  float*  sL   = (float*)(sm + OFF_SL);

  const __bf16* wqb = wsb;
  const __bf16* wkb = wsb + 16384;
  const __bf16* wgb = wsb + 32768;

  const int tid  = threadIdx.x;
  const int wave = tid >> 5;
  const int lane = tid & 31;
  const int half = lane >> 4;
  const int l    = lane & 15;
  const int b0   = blockIdx.x * 16;

  // ---------- load tile ----------
  {
    const float* gp = neigh + (size_t)blockIdx.x * 65536;  // 256 rows x 256
    #pragma unroll 8
    for (int i = 0; i < 64; ++i) {
      int idx = (i * 256 + tid) * 4;
      v4f v = *(const v4f*)(gp + idx);
      int row = idx >> 8, col = idx & 255;
      *(v4bf*)(sN + row * SN + col) = __builtin_convertvector(v, v4bf);
    }
    const float* cp = center + (size_t)b0 * 256;
    #pragma unroll
    for (int i = 0; i < 4; ++i) {
      int idx = (i * 256 + tid) * 4;
      v4f v = *(const v4f*)(cp + idx);
      int row = idx >> 8, col = idx & 255;
      *(v4f*)(sC + row * SC + col) = v;
      *(v4bf*)(sCb + row * SN + col) = __builtin_convertvector(v, v4bf);
    }
  }
  __syncthreads();

  // ---------- k projection: [256x256] x Wk^T[256x64] ----------
  {
    const int nt = wave & 3;          // N tile (16 cols of A-dim)
    const int mb = (wave >> 2) * 8;   // first of 8 M tiles
    v8f acc[8];
    #pragma unroll
    for (int m = 0; m < 8; ++m) acc[m] = v8f{};
    #pragma unroll
    for (int ks = 0; ks < 8; ++ks) {
      const int kb = ks * 32;
      const __bf16* bp = wkb + (nt * 16 + l) * 256 + kb + half * 16;
      v16bf bf = jn(ldb8(bp), ldb8(bp + 8));
      #pragma unroll
      for (int m = 0; m < 8; ++m) {
        const __bf16* ap = sN + ((mb + m) * 16 + l) * SN + kb + half * 16;
        v16bf af = jn(ldb8(ap), ldb8(ap + 8));
        acc[m] = BF16WMMA(af, bf, acc[m]);
      }
    }
    const int acol = nt * 16 + l;
    const float bkv = bk[acol];
    #pragma unroll
    for (int m = 0; m < 8; ++m) {
      #pragma unroll
      for (int r = 0; r < 8; ++r)
        sK[((mb + m) * 16 + r + half * 8) * SKsz + acol] = acc[m][r] + bkv;
    }
  }

  // ---------- q projection (waves 0..3): [16x256] x Wq^T[256x64] ----------
  if (wave < 4) {
    const int nt = wave;
    v8f acc = v8f{};
    #pragma unroll
    for (int ks = 0; ks < 8; ++ks) {
      const int kb = ks * 32;
      const __bf16* bp = wqb + (nt * 16 + l) * 256 + kb + half * 16;
      const __bf16* ap = sCb + l * SN + kb + half * 16;
      acc = BF16WMMA(jn(ldb8(ap), ldb8(ap + 8)), jn(ldb8(bp), ldb8(bp + 8)), acc);
    }
    const int acol = nt * 16 + l;
    const float bqv = bq[acol];
    #pragma unroll
    for (int r = 0; r < 8; ++r)
      sQ[(r + half * 8) * SKsz + acol] = acc[r] + bqv;
  }
  __syncthreads();

  // ---------- logits ----------
  {
    const int b = tid >> 4, kk = tid & 15;
    float s = 0.f;
    #pragma unroll
    for (int a = 0; a < 64; ++a) s += sQ[b * SKsz + a] * sK[tid * SKsz + a];
    sL[b * SL + kk] = s * 0.125f;  // SCALE = 64^-0.5
  }
  __syncthreads();

  // ---------- softmax + confidence reweight + renorm ----------
  float attn_val;
  {
    const int b = tid >> 4, kk = tid & 15;
    float lv[16], wv[16];
    #pragma unroll
    for (int j = 0; j < 16; ++j) {
      lv[j] = sL[b * SL + j];
      wv[j] = nwt[(size_t)(b0 + b) * 16 + j];
    }
    float mx = lv[0];
    #pragma unroll
    for (int j = 1; j < 16; ++j) mx = fmaxf(mx, lv[j]);
    float P = 0.f, S = 0.f;
    #pragma unroll
    for (int j = 0; j < 16; ++j) {
      float p = expf(lv[j] - mx);
      P += p;
      S += p * wv[j];
    }
    float pk = expf(lv[kk] - mx);
    // ref: (p/P*w) / (sum(p/P*w) + 1e-8) == p*w / (S + 1e-8*P)
    attn_val = pk * wv[kk] / (S + 1e-8f * P);
  }
  __syncthreads();
  sL[(tid >> 4) * SL + (tid & 15)] = attn_val;
  __syncthreads();

  // ---------- context = attn . neigh ----------
  {
    const int b = tid >> 4, dg = tid & 15;
    float ca[16];
    #pragma unroll
    for (int j = 0; j < 16; ++j) ca[j] = 0.f;
    #pragma unroll
    for (int kk = 0; kk < 16; ++kk) {
      float w = sL[b * SL + kk];
      const __bf16* p = sN + (b * 16 + kk) * SN + dg * 16;
      v8bf n0 = ldb8(p), n1 = ldb8(p + 8);
      #pragma unroll
      for (int j = 0; j < 8; ++j) {
        ca[j]     += w * (float)n0[j];
        ca[8 + j] += w * (float)n1[j];
      }
    }
    #pragma unroll
    for (int j = 0; j < 16; ++j) sCx[b * SC + dg * 16 + j] = ca[j];
    v8bf c0, c1;
    #pragma unroll
    for (int j = 0; j < 8; ++j) { c0[j] = (__bf16)ca[j]; c1[j] = (__bf16)ca[8 + j]; }
    *(v8bf*)(sCxb + b * SN + dg * 16)     = c0;
    *(v8bf*)(sCxb + b * SN + dg * 16 + 8) = c1;
  }
  __syncthreads();

  // ---------- gate GEMM: [16x512] x Wg^T[512x256], blend, store ----------
  {
    v8f g0 = v8f{}, g1 = v8f{};
    #pragma unroll
    for (int ks = 0; ks < 16; ++ks) {
      const int kb = ks * 32;
      const __bf16* ap = (kb < 256) ? (sCb + l * SN + kb + half * 16)
                                    : (sCxb + l * SN + (kb - 256) + half * 16);
      v16bf af = jn(ldb8(ap), ldb8(ap + 8));
      const __bf16* bp0 = wgb + ((2 * wave)     * 16 + l) * 512 + kb + half * 16;
      const __bf16* bp1 = wgb + ((2 * wave + 1) * 16 + l) * 512 + kb + half * 16;
      g0 = BF16WMMA(af, jn(ldb8(bp0), ldb8(bp0 + 8)), g0);
      g1 = BF16WMMA(af, jn(ldb8(bp1), ldb8(bp1 + 8)), g1);
    }
    #pragma unroll
    for (int i = 0; i < 2; ++i) {
      v8f g = i ? g1 : g0;
      const int n = (2 * wave + i) * 16 + l;
      const float bgv = bg[n];
      #pragma unroll
      for (int r = 0; r < 8; ++r) {
        const int M = r + half * 8;
        float z = g[r] + bgv;
        float gate = 1.f / (1.f + expf(-z));
        float ce = sC[M * SC + n];
        float cx = sCx[M * SC + n];
        float fused = gate * ce + (1.f - gate) * cx;
        out[(size_t)(b0 + M) * 256 + n] = vmask[b0 + M] ? fused : ce;
      }
    }
  }
}

extern "C" void kernel_launch(void* const* d_in, const int* in_sizes, int n_in,
                              void* d_out, int out_size, void* d_ws, size_t ws_size,
                              hipStream_t stream) {
  (void)in_sizes; (void)n_in; (void)out_size; (void)ws_size;
  const float* center = (const float*)d_in[0];
  const float* neigh  = (const float*)d_in[1];
  const float* nwt    = (const float*)d_in[2];
  const int*   vmask  = (const int*)d_in[3];
  const float* Wq     = (const float*)d_in[4];
  const float* bq     = (const float*)d_in[5];
  const float* Wk     = (const float*)d_in[6];
  const float* bk     = (const float*)d_in[7];
  const float* Wg     = (const float*)d_in[8];
  const float* bg     = (const float*)d_in[9];
  __bf16* wsb = (__bf16*)d_ws;

  na_weights_to_bf16<<<640, 256, 0, stream>>>(Wq, Wk, Wg, wsb);
  na_fused<<<2048, 256, SMEM_BYTES, stream>>>(center, neigh, nwt, vmask,
                                              bq, bk, bg, wsb, (float*)d_out);
}
